// GeneralConv2d_2662879723627
// MI455X (gfx1250) — compile-verified, matrix operand order checked by reading, not run on hardware
//
#include <hip/hip_runtime.h>

typedef __attribute__((ext_vector_type(16))) _Float16 v16h;
typedef __attribute__((ext_vector_type(8)))  _Float16 v8h;
typedef __attribute__((ext_vector_type(8)))  float    v8f;
typedef __attribute__((ext_vector_type(4)))  unsigned int u32x4;
typedef __attribute__((ext_vector_type(8)))  int      i32x8;
typedef __attribute__((ext_vector_type(4)))  int      i32x4;

#define OUT_C 256
#define IN_C  256
#define NB    32
#define HW    56
#define SP    (HW * HW)        // 3136
#define KTOT  (IN_C * 9)       // 2304
#define NTOT  (NB * SP)        // 100352

#define TILE_M 64
#define TILE_N 128
#define LDSTR  40              // 32 k-halfs + 8 pad halfs (80B row, keeps 16B align)

// ---------------------------------------------------------------------------
// Prep: scale[oc] = mean |w|; Aq[oc][tap*256+ic] = sign(w) in f16
// ---------------------------------------------------------------------------
__global__ __launch_bounds__(256)
void binconv_prep_kernel(const float* __restrict__ w,
                         _Float16* __restrict__ Aq,
                         float* __restrict__ scale) {
    const int oc = blockIdx.x;
    const int ic = threadIdx.x;
    const float* wrow = w + ((size_t)oc * IN_C + ic) * 9;
    float asum = 0.f;
#pragma unroll
    for (int t = 0; t < 9; ++t) {
        float v = wrow[t];
        asum += fabsf(v);
        Aq[(size_t)oc * KTOT + t * IN_C + ic] =
            (v >= 0.f) ? (_Float16)1.0f : (_Float16)(-1.0f);
    }
    __shared__ float red[256];
    red[ic] = asum;
    __syncthreads();
    for (int s = 128; s > 0; s >>= 1) {
        if (ic < s) red[ic] += red[ic + s];
        __syncthreads();
    }
    if (ic == 0) scale[oc] = red[0] * (1.0f / (float)KTOT);
}

// ---------------------------------------------------------------------------
// One-time transform: x f32 NCHW -> xT f16 NHWC (channels contiguous)
// ---------------------------------------------------------------------------
__global__ __launch_bounds__(256)
void binconv_xpose_kernel(const float* __restrict__ x, _Float16* __restrict__ xT) {
    const int t   = blockIdx.x * 256 + threadIdx.x;  // one thread per 8 channels
    const int spc = t >> 5;                          // spatial index (b*SP + s2)
    const int c0  = (t & 31) * 8;
    const int b   = spc / SP;
    const int s2  = spc % SP;
    _Float16 tmp[8];
#pragma unroll
    for (int i = 0; i < 8; ++i)
        tmp[i] = (_Float16)x[((size_t)b * IN_C + c0 + i) * SP + s2];
    *(uint4*)&xT[(size_t)spc * IN_C + c0] = *(uint4*)tmp;
}

// ---------------------------------------------------------------------------
// TDM: 2D tile load Aq[64 rows x 32 halfs] -> LDS with 16B pad after each 64B row
// ---------------------------------------------------------------------------
__device__ __forceinline__
void tdm_load_A_tile(unsigned lds_addr, unsigned long long gaddr) {
    u32x4 g0;
    g0[0] = 1u;                                  // count=1, user mode
    g0[1] = lds_addr;                            // lds_addr (bytes)
    g0[2] = (unsigned)(gaddr & 0xFFFFFFFFull);   // global_addr[31:0]
    g0[3] = (unsigned)((gaddr >> 32) & 0x1FFFFFFull) | (2u << 30); // [56:32] | type=2

    i32x8 g1;
    // data_size=1 (2B) | pad_enable | pad_interval=3 (16 DW) | pad_amount=3 (4 DW)
    g1[0] = (1 << 16) | (1 << 20) | (3 << 22) | (3 << 25);
    g1[1] = (32 << 16);            // tensor_dim0 = 32 (low half at bits 63:48)
    g1[2] = (64 << 16);            // tensor_dim1 = 64 (low half)
    g1[3] = (32 << 16);            // tile_dim0 = 32
    g1[4] = 64;                    // tile_dim1 = 64, tile_dim2 = 0
    g1[5] = KTOT;                  // tensor_dim0_stride = 2304 elements
    g1[6] = 0;
    g1[7] = 0;

    i32x4 gz = {0, 0, 0, 0};
#if defined(__clang_major__) && (__clang_major__ >= 23)
    i32x8 gz8 = {0, 0, 0, 0, 0, 0, 0, 0};
    __builtin_amdgcn_tensor_load_to_lds(g0, g1, gz, gz, gz8, 0);
#else
    __builtin_amdgcn_tensor_load_to_lds(g0, g1, gz, gz, 0);
#endif
}

// ---------------------------------------------------------------------------
// Shared compute core: fragments + 2x4 WMMA + epilogue pieces
// ---------------------------------------------------------------------------
__device__ __forceinline__
void wmma_2x4(const _Float16* lA, const _Float16* lB,
              int wm, int wn, int lane, v8f acc[2][4]) {
    const int h = lane >> 4;
    const int r = lane & 15;
    v16h afr[2], bfr[4];
#pragma unroll
    for (int t = 0; t < 2; ++t) {
        const int row = wm * 32 + t * 16 + r;
        v8h alo = *(const v8h*)&lA[row * LDSTR + 8 * h];
        v8h ahi = *(const v8h*)&lA[row * LDSTR + 16 + 8 * h];
        v16h av;
#pragma unroll
        for (int i = 0; i < 8; ++i) { av[i] = alo[i]; av[8 + i] = ahi[i]; }
        afr[t] = av;
    }
#pragma unroll
    for (int t = 0; t < 4; ++t) {
        const int col = wn * 64 + t * 16 + r;
        v8h blo = *(const v8h*)&lB[col * LDSTR + 8 * h];
        v8h bhi = *(const v8h*)&lB[col * LDSTR + 16 + 8 * h];
        v16h bv;
#pragma unroll
        for (int i = 0; i < 8; ++i) { bv[i] = blo[i]; bv[8 + i] = bhi[i]; }
        bfr[t] = bv;
    }
#pragma unroll
    for (int tm = 0; tm < 2; ++tm)
#pragma unroll
        for (int tn = 0; tn < 4; ++tn)
            acc[tm][tn] = __builtin_amdgcn_wmma_f32_16x16x32_f16(
                false, afr[tm], false, bfr[tn],
                (short)0, acc[tm][tn], false, false);
}

__device__ __forceinline__
void epilogue_store(float* __restrict__ y, const float* __restrict__ scale,
                    int mBase, int nBase, int wm, int wn, int lane, v8f acc[2][4]) {
    const int h  = lane >> 4;
    const int nl = lane & 15;
#pragma unroll
    for (int tm = 0; tm < 2; ++tm) {
#pragma unroll
        for (int tn = 0; tn < 4; ++tn) {
            const int gnn = nBase + wn * 64 + tn * 16 + nl;
            const int ob  = gnn / SP;
            const int osp = gnn % SP;
#pragma unroll
            for (int rr = 0; rr < 8; ++rr) {
                const int oc = mBase + wm * 32 + tm * 16 + rr + 8 * h;
                y[((size_t)ob * OUT_C + oc) * SP + osp] = acc[tm][tn][rr] * scale[oc];
            }
        }
    }
}

// ---------------------------------------------------------------------------
// Fast path: NHWC f16 input; B tile staged with GLOBAL_LOAD_ASYNC_TO_LDS_B128,
// A tile staged with TDM. 4 waves; wave = 32x64 quadrant of 64x128 block tile.
// ---------------------------------------------------------------------------
__global__ __launch_bounds__(128)
void binconv_wmma_async_kernel(const _Float16* __restrict__ xT,
                               const _Float16* __restrict__ Aq,
                               const float* __restrict__ scale,
                               float* __restrict__ y) {
    __shared__ __align__(16) _Float16 lA[TILE_M * LDSTR];
    __shared__ __align__(16) _Float16 lB[TILE_N * LDSTR];

    const int tid  = threadIdx.x;
    const int lane = tid & 31;
    const int wave = tid >> 5;
    const int wm   = wave >> 1;
    const int wn   = wave & 1;

    const int nBase = blockIdx.x * TILE_N;   // 784 exact
    const int mBase = blockIdx.y * TILE_M;   // 4 exact

    // B loader: thread owns one of 128 n-rows
    const int gn = nBase + tid;
    const int b  = gn / SP;
    const int s2 = gn % SP;
    const int oh = s2 / HW;
    const int ow = s2 % HW;

    const unsigned ldsA  = (unsigned)(size_t)(&lA[0]);
    const unsigned ldsBr = (unsigned)(size_t)(&lB[0]) + (unsigned)tid * (LDSTR * 2);
    const unsigned long long aqBase =
        (unsigned long long)(size_t)(Aq + (size_t)mBase * KTOT);

    v8f acc[2][4] = {};

    for (int tap = 0; tap < 9; ++tap) {
        const int kh = tap / 3, kw = tap % 3;
        const int ih = oh + kh - 1;
        const int iw = ow + kw - 1;
        const bool inb = (ih >= 0) && (ih < HW) && (iw >= 0) && (iw < HW);
        const _Float16* srcRow =
            xT + ((size_t)b * SP + (size_t)ih * HW + iw) * IN_C;   // deref only if inb

        for (int icb = 0; icb < IN_C; icb += 32) {
            // ---- A tile via Tensor Data Mover (one wave issues; EXEC ignored) ----
            if (wave == 0)
                tdm_load_A_tile(ldsA,
                                aqBase + (unsigned long long)(tap * IN_C + icb) * 2ull);
            // ---- B row: 64B contiguous -> 4x async b128 straight into LDS ----
            if (inb) {
                const unsigned long long g =
                    (unsigned long long)(size_t)(srcRow + icb);
                asm volatile("global_load_async_to_lds_b128 %0, %1, off"
                             :: "v"(ldsBr), "v"(g) : "memory");
                asm volatile("global_load_async_to_lds_b128 %0, %1, off offset:16"
                             :: "v"(ldsBr), "v"(g) : "memory");
                asm volatile("global_load_async_to_lds_b128 %0, %1, off offset:32"
                             :: "v"(ldsBr), "v"(g) : "memory");
                asm volatile("global_load_async_to_lds_b128 %0, %1, off offset:48"
                             :: "v"(ldsBr), "v"(g) : "memory");
                if (icb + 32 < IN_C)
                    __builtin_prefetch(srcRow + icb + 32, 0, 1);   // global_prefetch_b8
            } else {
                const uint4 z = {0u, 0u, 0u, 0u};
#pragma unroll
                for (int s = 0; s < 4; ++s)
                    *(uint4*)&lB[tid * LDSTR + s * 8] = z;
            }
            asm volatile("s_wait_asynccnt 0" ::: "memory");
            if (wave == 0) __builtin_amdgcn_s_wait_tensorcnt(0);
            __syncthreads();

            wmma_2x4(lA, lB, wm, wn, lane, acc);

            __syncthreads();
        }
    }
    epilogue_store(y, scale, mBase, nBase, wm, wn, lane, acc);
}

// ---------------------------------------------------------------------------
// Fallback path (small workspace): gather f32 + cvt.
// ---------------------------------------------------------------------------
__global__ __launch_bounds__(128)
void binconv_wmma_gather_kernel(const float* __restrict__ x,
                                const _Float16* __restrict__ Aq,
                                const float* __restrict__ scale,
                                float* __restrict__ y) {
    __shared__ __align__(16) _Float16 lA[TILE_M * LDSTR];
    __shared__ __align__(16) _Float16 lB[TILE_N * LDSTR];

    const int tid  = threadIdx.x;
    const int lane = tid & 31;
    const int wave = tid >> 5;
    const int wm   = wave >> 1;
    const int wn   = wave & 1;

    const int nBase = blockIdx.x * TILE_N;
    const int mBase = blockIdx.y * TILE_M;

    const int r0    = tid >> 1;
    const int lhalf = (tid & 1) * 16;

    int bb[2], oh[2], ow[2];
#pragma unroll
    for (int j = 0; j < 2; ++j) {
        const int gn = nBase + r0 + 64 * j;
        bb[j] = gn / SP;
        const int sprem = gn % SP;
        oh[j] = sprem / HW;
        ow[j] = sprem % HW;
    }

    const unsigned ldsA = (unsigned)(size_t)(&lA[0]);
    const unsigned long long aqBase =
        (unsigned long long)(size_t)(Aq + (size_t)mBase * KTOT);

    v8f acc[2][4] = {};

    for (int tap = 0; tap < 9; ++tap) {
        const int kh = tap / 3, kw = tap % 3;
        long xoff[2];
        bool inb[2];
#pragma unroll
        for (int j = 0; j < 2; ++j) {
            const int ih = oh[j] + kh - 1;
            const int iw = ow[j] + kw - 1;
            inb[j]  = (ih >= 0) && (ih < HW) && (iw >= 0) && (iw < HW);
            xoff[j] = (long)bb[j] * IN_C * SP + (long)ih * HW + iw;
        }

        for (int icb = 0; icb < IN_C; icb += 32) {
            if (wave == 0)
                tdm_load_A_tile(ldsA,
                                aqBase + (unsigned long long)(tap * IN_C + icb) * 2ull);
#pragma unroll
            for (int j = 0; j < 2; ++j) {
                _Float16 tmp[16];
#pragma unroll
                for (int i = 0; i < 16; ++i) {
                    const int ic = icb + lhalf + i;
                    float v = inb[j] ? x[xoff[j] + (long)ic * SP] : 0.f;
                    tmp[i] = (_Float16)v;
                }
                const int row = r0 + 64 * j;
                *(uint4*)&lB[row * LDSTR + lhalf]     = *(uint4*)&tmp[0];
                *(uint4*)&lB[row * LDSTR + lhalf + 8] = *(uint4*)&tmp[8];
                if (inb[j] && (icb + 32 < IN_C))
                    __builtin_prefetch(&x[xoff[j] + (long)(icb + 32 + lhalf) * SP], 0, 1);
            }
            if (wave == 0) __builtin_amdgcn_s_wait_tensorcnt(0);
            __syncthreads();

            wmma_2x4(lA, lB, wm, wn, lane, acc);

            __syncthreads();
        }
    }
    epilogue_store(y, scale, mBase, nBase, wm, wn, lane, acc);
}

// ---------------------------------------------------------------------------
extern "C" void kernel_launch(void* const* d_in, const int* in_sizes, int n_in,
                              void* d_out, int out_size, void* d_ws, size_t ws_size,
                              hipStream_t stream) {
    const float* x = (const float*)d_in[0];   // (32,256,56,56) f32
    const float* w = (const float*)d_in[1];   // (256*256*9, 1) f32
    float* y = (float*)d_out;                 // (32,256,56,56) f32

    const size_t aqBytes = (size_t)OUT_C * KTOT * sizeof(_Float16);  // 1.125 MB
    const size_t scBytes = 1024;                                     // 256 f32, padded
    const size_t xtBytes = (size_t)NTOT * IN_C * sizeof(_Float16);   // 51.4 MB

    _Float16* Aq   = (_Float16*)d_ws;
    float*    scal = (float*)((char*)d_ws + aqBytes);
    _Float16* xT   = (_Float16*)((char*)d_ws + aqBytes + scBytes);

    binconv_prep_kernel<<<OUT_C, 256, 0, stream>>>(w, Aq, scal);

    dim3 grid(NTOT / TILE_N, OUT_C / TILE_M);  // 784 x 4, exact

    if (ws_size >= aqBytes + scBytes + xtBytes) {
        // fast path: NHWC f16 staged once, hot loop uses async-to-LDS + TDM
        binconv_xpose_kernel<<<(NTOT * (IN_C / 8)) / 256, 256, 0, stream>>>(x, xT);
        binconv_wmma_async_kernel<<<grid, 128, 0, stream>>>(xT, Aq, scal, y);
    } else {
        binconv_wmma_gather_kernel<<<grid, 128, 0, stream>>>(x, Aq, scal, y);
    }
}
